// MultiHeadAttention_49039936585736
// MI455X (gfx1250) — matrix operand order repaired and checked by reading, not compiled, not run on hardware
//
#include <hip/hip_runtime.h>
#include <cstdint>

#define DEVI __device__ __forceinline__

typedef _Float16 v16h __attribute__((ext_vector_type(16)));
typedef _Float16 v8h  __attribute__((ext_vector_type(8)));
typedef float    v8f  __attribute__((ext_vector_type(8)));
typedef unsigned int v4u __attribute__((ext_vector_type(4)));
typedef int      v8i  __attribute__((ext_vector_type(8)));
typedef int      v4i  __attribute__((ext_vector_type(4)));

#if defined(__has_builtin)
#if __has_builtin(__builtin_amdgcn_tensor_load_to_lds) && \
    __has_builtin(__builtin_amdgcn_s_wait_tensorcnt)
#define HAS_TDM 1
#endif
#endif

namespace {
constexpr int Bb   = 4;
constexpr int Tt   = 2048;
constexpr int Ee   = 1024;
constexpr int Hh   = 16;
constexpr int Dd   = 64;
constexpr int Rows = Bb * Tt;   // 8192
constexpr int E3   = 3 * Ee;    // 3072
}

// ---------------------------------------------------------------- WMMA core
DEVI v8f wmma_f16(v16h a, v16h b, v8f c) {
  // D = A(16x32 f16) * B(32x16 f16) + C(16x16 f32)
  return __builtin_amdgcn_wmma_f32_16x16x32_f16(false, a, false, b, (short)0, c,
                                                false, false);
}

// A-fragment (16x32, f16): lane (l&15)=row M; halves 0..7 = K (l>>4)*8+{0..7},
// halves 8..15 = K (l>>4)*8+16+{0..7}.  (ISA 7.12.2 dense A table)
DEVI v16h load_frag_a(const _Float16* tile, int stride, int rowBase, int kBase) {
  const int l = threadIdx.x & 31;
  const _Float16* p =
      tile + (size_t)(rowBase + (l & 15)) * stride + kBase + ((l >> 4) << 3);
  v8h lo = *(const v8h*)p;
  v8h hi = *(const v8h*)(p + 16);
  v16h r;
#pragma unroll
  for (int i = 0; i < 8; ++i) { r[i] = lo[i]; r[i + 8] = hi[i]; }
  return r;
}

// B-fragment (32x16, f16): lane (l&15)=column N; halves 0..7 = K (l>>4)*8+{0..7},
// halves 8..15 = K (l>>4)*8+16+{0..7}.
// Source tile stores B column-major (row = output column, contiguous in K).
DEVI v16h load_frag_b(const _Float16* tile, int stride, int colBase, int kBase) {
  const int l = threadIdx.x & 31;
  const _Float16* p =
      tile + (size_t)(colBase + (l & 15)) * stride + kBase + ((l >> 4) << 3);
  v8h lo = *(const v8h*)p;
  v8h hi = *(const v8h*)(p + 16);
  v16h r;
#pragma unroll
  for (int i = 0; i < 8; ++i) { r[i] = lo[i]; r[i + 8] = hi[i]; }
  return r;
}

// Row reductions across the 16 lanes sharing one C row (xor masks 1..8 only).
DEVI float red_max16(float v) {
#pragma unroll
  for (int m = 1; m < 16; m <<= 1) v = fmaxf(v, __shfl_xor(v, m, 32));
  return v;
}
DEVI float red_sum16(float v) {
#pragma unroll
  for (int m = 1; m < 16; m <<= 1) v += __shfl_xor(v, m, 32);
  return v;
}

// --------------------------------------------- CDNA5 async global->LDS copy
// Per-lane 16B DMA into LDS, tracked by ASYNCcnt (no VGPR data movement).
DEVI void async_b128(const void* gptr, void* ldsptr) {
  unsigned lds = (unsigned)(uintptr_t)ldsptr;        // low 32b = LDS offset
  unsigned long long ga = (unsigned long long)(uintptr_t)gptr;
  asm volatile("global_load_async_to_lds_b128 %0, %1, off"
               :: "v"(lds), "v"(ga)
               : "memory");
}
DEVI void wait_async0() {
#if defined(__has_builtin)
#if __has_builtin(__builtin_amdgcn_s_wait_asynccnt)
  __builtin_amdgcn_s_wait_asynccnt(0);
#else
  asm volatile("s_wait_asynccnt 0" ::: "memory");
#endif
#else
  asm volatile("s_wait_asynccnt 0" ::: "memory");
#endif
}

#ifdef HAS_TDM
// ------------------------------------------------- Tensor Data Mover (TDM)
// 2D tile load: tileW=64 f16 elements per row, tileH=64 rows, row stride
// `strideElems`, destination LDS rows padded by 16B after each 128B row
// (pad_interval code 4 = 32 DWORDs, pad_amount code 3 = 4 DWORDs) so the
// LDS stride is 72 halves = LDK.  D# layout per CDNA5 ISA ch.8.
// This toolchain's builtin takes 6 args: (g0, g1, g2, g3, extra v8i, cpol).
DEVI void tdm_load_tile64(const void* gptr, unsigned ldsAddr, int strideElems) {
  unsigned long long ga = (unsigned long long)(uintptr_t)gptr;
  v4u g0;
  g0[0] = 1u;                                     // count=1, user descriptor
  g0[1] = ldsAddr;                                // lds_addr
  g0[2] = (unsigned)ga;                           // global_addr[31:0]
  g0[3] = (unsigned)(ga >> 32) | (2u << 30);      // global_addr[56:32]|type=2
  const int tensorD0 = strideElems;               // >= tile width (no OOB)
  const int tensorD1 = 1 << 20;
  v8i g1;
  g1[0] = (1 << 16) | (1 << 20) | (4 << 22) | (3 << 25);  // 2B, pad 4dw/32dw
  g1[1] = (tensorD0 & 0xFFFF) << 16;                       // abar=0, dim0 lo
  g1[2] = (tensorD0 >> 16) | ((tensorD1 & 0xFFFF) << 16);
  g1[3] = (tensorD1 >> 16) | (64 << 16);                   // tile_dim0 = 64
  g1[4] = 64;                                              // tile_dim1 = 64
  g1[5] = strideElems;                                     // dim0_stride lo
  g1[6] = 0;
  g1[7] = 0;                                               // dim1_stride = 0
  v4i g2; g2[0] = 0; g2[1] = 0; g2[2] = 0; g2[3] = 0;      // 2D: unused
  v4i g3; g3[0] = 0; g3[1] = 0; g3[2] = 0; g3[3] = 0;
  v8i gx;
#pragma unroll
  for (int i = 0; i < 8; ++i) gx[i] = 0;
  __builtin_amdgcn_tensor_load_to_lds(g0, g1, g2, g3, gx, 0);
}
#endif

// ------------------------------------------------------------- prep kernels
__global__ void k_mask(const long long* __restrict__ ids, float* __restrict__ m,
                       int n) {
  int i = blockIdx.x * 256 + threadIdx.x;
  if (i < n) m[i] = (ids[i] == 0) ? -1e30f : 0.0f;
}

// wT[n][k] = (f16) w[k][n]
__global__ void k_wt(const float* __restrict__ w, _Float16* __restrict__ wT,
                     int K, int N) {
  int k = blockIdx.x * 256 + threadIdx.x;
  int n = blockIdx.y;
  if (k < K) wT[(size_t)n * K + k] = (_Float16)w[(size_t)k * N + n];
}

// -------------------------------------------------- QKV projection (GEMM 1)
__global__ __launch_bounds__(256) void k_qkv_gemm(
    const float* __restrict__ x, const _Float16* __restrict__ wT,
    _Float16* __restrict__ Qh, _Float16* __restrict__ Kh,
    _Float16* __restrict__ Vt) {
  constexpr int BM = 128, BN = 128, BK = 32, KD = Ee, LDT = 40,
                KSTEPS = KD / BK;
  __shared__ _Float16 sA[2][BM * LDT];
  __shared__ _Float16 sB[2][BN * LDT];
  const int tid = threadIdx.x, wid = tid >> 5, l = tid & 31;
  const int bm = blockIdx.x, bn = blockIdx.y;
  const int waveM = wid >> 2, waveN = wid & 3;  // 2 x 4 wave grid

  v8f acc[4][2];
#pragma unroll
  for (int mi = 0; mi < 4; ++mi)
#pragma unroll
    for (int ni = 0; ni < 2; ++ni)
#pragma unroll
      for (int r = 0; r < 8; ++r) acc[mi][ni][r] = 0.0f;

  auto stage = [&](int ks, int buf) {
#pragma unroll
    for (int i = 0; i < 2; ++i) {          // A: 128 rows x 32 halves (cvt f32)
      int chunk = tid + i * 256;
      int row = chunk >> 2, cc = chunk & 3;
      const float4* gp =
          (const float4*)(x + (size_t)(bm * BM + row) * KD + ks * BK + cc * 8);
      float4 f0 = gp[0], f1 = gp[1];
      v8h hv;
      hv[0] = (_Float16)f0.x; hv[1] = (_Float16)f0.y;
      hv[2] = (_Float16)f0.z; hv[3] = (_Float16)f0.w;
      hv[4] = (_Float16)f1.x; hv[5] = (_Float16)f1.y;
      hv[6] = (_Float16)f1.z; hv[7] = (_Float16)f1.w;
      *(v8h*)&sA[buf][row * LDT + cc * 8] = hv;
    }
#pragma unroll
    for (int i = 0; i < 2; ++i) {          // B: pure f16 copy -> async DMA
      int chunk = tid + i * 256;
      int row = chunk >> 2, cc = chunk & 3;
      async_b128(wT + (size_t)(bn * BN + row) * KD + ks * BK + cc * 8,
                 &sB[buf][row * LDT + cc * 8]);
    }
  };

  stage(0, 0);
  wait_async0();
  __syncthreads();
  for (int ks = 0; ks < KSTEPS; ++ks) {
    const int cur = ks & 1;
    if (ks + 1 < KSTEPS) stage(ks + 1, cur ^ 1);
    if (ks + 2 < KSTEPS) {
      __builtin_prefetch(x + (size_t)(bm * BM + (tid >> 2)) * KD + (ks + 2) * BK, 0, 0);
      __builtin_prefetch(wT + (size_t)(bn * BN + (tid >> 2)) * KD + (ks + 2) * BK, 0, 0);
    }
    v16h af[4], bf[2];
#pragma unroll
    for (int mi = 0; mi < 4; ++mi)
      af[mi] = load_frag_a(sA[cur], LDT, waveM * 64 + mi * 16, 0);
#pragma unroll
    for (int ni = 0; ni < 2; ++ni)
      bf[ni] = load_frag_b(sB[cur], LDT, waveN * 32 + ni * 16, 0);
#pragma unroll
    for (int mi = 0; mi < 4; ++mi)
#pragma unroll
      for (int ni = 0; ni < 2; ++ni)
        acc[mi][ni] = wmma_f16(af[mi], bf[ni], acc[mi][ni]);
    wait_async0();
    __syncthreads();
  }

  // scatter epilogue: decode global column into (q|k|v, head, d)
#pragma unroll
  for (int mi = 0; mi < 4; ++mi)
#pragma unroll
    for (int ni = 0; ni < 2; ++ni)
#pragma unroll
      for (int r = 0; r < 8; ++r) {
        int gRow = bm * BM + waveM * 64 + mi * 16 + r + ((l >> 4) << 3);
        int gCol = bn * BN + waveN * 32 + ni * 16 + (l & 15);
        float v = acc[mi][ni][r];
        int which = gCol >> 10, e = gCol & 1023;
        int h = e >> 6, d = e & 63;
        int b = gRow >> 11, t = gRow & 2047;
        if (which == 0)
          Qh[((size_t)(b * Hh + h) * Tt + t) * Dd + d] = (_Float16)v;
        else if (which == 1)
          Kh[((size_t)(b * Hh + h) * Tt + t) * Dd + d] = (_Float16)v;
        else
          Vt[((size_t)(b * Hh + h) * Dd + d) * Tt + t] = (_Float16)v;
      }
}

// -------------------------------------------------------- flash attention
__global__ __launch_bounds__(256) void k_attn(
    const _Float16* __restrict__ Qh, const _Float16* __restrict__ Kh,
    const _Float16* __restrict__ Vt, const float* __restrict__ maskAdd,
    _Float16* __restrict__ Og) {
  constexpr int LDK = 72;  // 144B rows: matches TDM pad (128B data + 16B pad)
  __shared__ _Float16 sK[64 * LDK];      // [key][d]
  __shared__ _Float16 sV[64 * LDK];      // [d][key]  (V pre-transposed)
  __shared__ _Float16 sP[8][16 * LDK];   // per-wave P staging [qrow][key]
  __shared__ float sM[64];
  const int tid = threadIdx.x, wid = tid >> 5, l = tid & 31;
  const int bh = blockIdx.x;
  const int b = bh >> 4, h = bh & 15;
  const int qBase = blockIdx.y * 128;
  const int myQ = qBase + wid * 16;

  const _Float16* qp = Qh + ((size_t)bh * Tt + myQ) * Dd;
  const v16h qf0 = load_frag_a(qp, Dd, 0, 0);
  const v16h qf1 = load_frag_a(qp, Dd, 0, 32);

  v8f o[4];
#pragma unroll
  for (int n = 0; n < 4; ++n)
#pragma unroll
    for (int r = 0; r < 8; ++r) o[n][r] = 0.0f;
  float mrow[8], lrow[8];
#pragma unroll
  for (int r = 0; r < 8; ++r) { mrow[r] = -1e30f; lrow[r] = 0.0f; }

  const int nkb = (qBase + 128) >> 6;  // causal: key blocks covering q tile
  for (int kb = 0; kb < nkb; ++kb) {
    const int kBase = kb * 64;
    __syncthreads();  // previous iteration done with sK/sV
#ifdef HAS_TDM
    if (wid == 0) {
      tdm_load_tile64(Kh + ((size_t)bh * Tt + kBase) * Dd,
                      (unsigned)(uintptr_t)(void*)sK, Dd);
      tdm_load_tile64(Vt + (size_t)bh * Dd * Tt + kBase,
                      (unsigned)(uintptr_t)(void*)sV, Tt);
    }
#else
#pragma unroll
    for (int i = 0; i < 2; ++i) {
      int chunk = tid + i * 256;
      int row = chunk >> 3, cc = chunk & 7;
      v8h kv = *(const v8h*)(Kh + ((size_t)bh * Tt + kBase + row) * Dd + cc * 8);
      *(v8h*)&sK[row * LDK + cc * 8] = kv;
      v8h vv = *(const v8h*)(Vt + ((size_t)bh * Dd + row) * Tt + kBase + cc * 8);
      *(v8h*)&sV[row * LDK + cc * 8] = vv;
    }
#endif
    if (tid < 64) sM[tid] = maskAdd[(size_t)b * Tt + kBase + tid];
#ifdef HAS_TDM
    if (wid == 0) __builtin_amdgcn_s_wait_tensorcnt(0);
#endif
    __syncthreads();

    // S = Q @ K^T  (4 key tiles x 2 K-chunks)
    float sc[4][8];
#pragma unroll
    for (int n = 0; n < 4; ++n) {
      v8f s;
#pragma unroll
      for (int r = 0; r < 8; ++r) s[r] = 0.0f;
      s = wmma_f16(qf0, load_frag_b(sK, LDK, n * 16, 0), s);
      s = wmma_f16(qf1, load_frag_b(sK, LDK, n * 16, 32), s);
#pragma unroll
      for (int r = 0; r < 8; ++r) sc[n][r] = s[r];
    }

    // scale + causal + pad mask (finite -1e30 avoids inf-inf NaNs)
    const int row0 = myQ + ((l >> 4) << 3);
#pragma unroll
    for (int n = 0; n < 4; ++n) {
      const int kIdx = kBase + n * 16 + (l & 15);
      const float ma = sM[n * 16 + (l & 15)];
#pragma unroll
      for (int r = 0; r < 8; ++r) {
        float sv = sc[n][r] * 0.125f + ma;  // 1/sqrt(64)
        sc[n][r] = (kIdx <= row0 + r) ? sv : -1e30f;
      }
    }

    // online softmax update
    float alpha[8];
#pragma unroll
    for (int r = 0; r < 8; ++r) {
      float mx = fmaxf(fmaxf(sc[0][r], sc[1][r]), fmaxf(sc[2][r], sc[3][r]));
      mx = red_max16(mx);
      float mnew = fmaxf(mrow[r], mx);
      alpha[r] = __expf(mrow[r] - mnew);
      mrow[r] = mnew;
      float rs = 0.0f;
#pragma unroll
      for (int n = 0; n < 4; ++n) {
        float e = __expf(sc[n][r] - mnew);
        sc[n][r] = e;
        rs += e;
      }
      rs = red_sum16(rs);
      lrow[r] = lrow[r] * alpha[r] + rs;
    }
#pragma unroll
    for (int n = 0; n < 4; ++n)
#pragma unroll
      for (int r = 0; r < 8; ++r) o[n][r] = o[n][r] * alpha[r];

    // C-layout -> A-layout via per-wave LDS round trip (LDS in-order per wave)
    _Float16* pw = sP[wid];
#pragma unroll
    for (int n = 0; n < 4; ++n)
#pragma unroll
      for (int r = 0; r < 8; ++r)
        pw[(size_t)(r + ((l >> 4) << 3)) * LDK + n * 16 + (l & 15)] =
            (_Float16)sc[n][r];

    const v16h pf0 = load_frag_a(pw, LDK, 0, 0);
    const v16h pf1 = load_frag_a(pw, LDK, 0, 32);
#pragma unroll
    for (int n = 0; n < 4; ++n) {
      o[n] = wmma_f16(pf0, load_frag_b(sV, LDK, n * 16, 0), o[n]);
      o[n] = wmma_f16(pf1, load_frag_b(sV, LDK, n * 16, 32), o[n]);
    }
  }

  // normalize and write head-merged f16 output Og[b, t, h*64 + d]
#pragma unroll
  for (int n = 0; n < 4; ++n)
#pragma unroll
    for (int r = 0; r < 8; ++r) {
      int q = myQ + ((l >> 4) << 3) + r;
      int d = n * 16 + (l & 15);
      float v = o[n][r] / lrow[r];
      Og[((size_t)b * Tt + q) * Ee + h * Dd + d] = (_Float16)v;
    }
}

// ------------------------------------------------ output projection (GEMM 2)
__global__ __launch_bounds__(256) void k_out_gemm(
    const _Float16* __restrict__ A, const _Float16* __restrict__ wT,
    const float* __restrict__ bias, float* __restrict__ out) {
  constexpr int BM = 128, BN = 128, BK = 32, KD = Ee, LDT = 40,
                KSTEPS = KD / BK;
  __shared__ _Float16 sA[2][BM * LDT];
  __shared__ _Float16 sB[2][BN * LDT];
  const int tid = threadIdx.x, wid = tid >> 5, l = tid & 31;
  const int bm = blockIdx.x, bn = blockIdx.y;
  const int waveM = wid >> 2, waveN = wid & 3;

  v8f acc[4][2];
#pragma unroll
  for (int mi = 0; mi < 4; ++mi)
#pragma unroll
    for (int ni = 0; ni < 2; ++ni)
#pragma unroll
      for (int r = 0; r < 8; ++r) acc[mi][ni][r] = 0.0f;

  auto stage = [&](int ks, int buf) {
#pragma unroll
    for (int i = 0; i < 2; ++i) {          // both operands f16 -> async DMA
      int chunk = tid + i * 256;
      int row = chunk >> 2, cc = chunk & 3;
      async_b128(A + (size_t)(bm * BM + row) * KD + ks * BK + cc * 8,
                 &sA[buf][row * LDT + cc * 8]);
      async_b128(wT + (size_t)(bn * BN + row) * KD + ks * BK + cc * 8,
                 &sB[buf][row * LDT + cc * 8]);
    }
  };

  stage(0, 0);
  wait_async0();
  __syncthreads();
  for (int ks = 0; ks < KSTEPS; ++ks) {
    const int cur = ks & 1;
    if (ks + 1 < KSTEPS) stage(ks + 1, cur ^ 1);
    if (ks + 2 < KSTEPS) {
      __builtin_prefetch(A + (size_t)(bm * BM + (tid >> 2)) * KD + (ks + 2) * BK, 0, 0);
      __builtin_prefetch(wT + (size_t)(bn * BN + (tid >> 2)) * KD + (ks + 2) * BK, 0, 0);
    }
    v16h af[4], bf[2];
#pragma unroll
    for (int mi = 0; mi < 4; ++mi)
      af[mi] = load_frag_a(sA[cur], LDT, waveM * 64 + mi * 16, 0);
#pragma unroll
    for (int ni = 0; ni < 2; ++ni)
      bf[ni] = load_frag_b(sB[cur], LDT, waveN * 32 + ni * 16, 0);
#pragma unroll
    for (int mi = 0; mi < 4; ++mi)
#pragma unroll
      for (int ni = 0; ni < 2; ++ni)
        acc[mi][ni] = wmma_f16(af[mi], bf[ni], acc[mi][ni]);
    wait_async0();
    __syncthreads();
  }

#pragma unroll
  for (int mi = 0; mi < 4; ++mi)
#pragma unroll
    for (int ni = 0; ni < 2; ++ni)
#pragma unroll
      for (int r = 0; r < 8; ++r) {
        int gRow = bm * BM + waveM * 64 + mi * 16 + r + ((l >> 4) << 3);
        int gCol = bn * BN + waveN * 32 + ni * 16 + (l & 15);
        out[(size_t)gRow * Ee + gCol] = acc[mi][ni][r] + bias[gCol];
      }
}

// --------------------------------------------------------------- launcher
extern "C" void kernel_launch(void* const* d_in, const int* in_sizes, int n_in,
                              void* d_out, int out_size, void* d_ws,
                              size_t ws_size, hipStream_t stream) {
  (void)in_sizes; (void)n_in; (void)out_size; (void)ws_size;
  const float*     x     = (const float*)d_in[0];
  const long long* ids   = (const long long*)d_in[1];  // int64 input_ids
  const float*     w_qkv = (const float*)d_in[2];
  const float*     w_out = (const float*)d_in[3];
  const float*     b_out = (const float*)d_in[4];
  float* out = (float*)d_out;

  char* p = (char*)d_ws;
  auto take = [&](size_t bytes) {
    char* r = p;
    p += (bytes + 255) & ~(size_t)255;
    return r;
  };
  _Float16* wqkvT = (_Float16*)take((size_t)E3 * Ee * 2);   //  6 MB
  _Float16* woutT = (_Float16*)take((size_t)Ee * Ee * 2);   //  2 MB
  _Float16* Qh    = (_Float16*)take((size_t)Rows * Ee * 2); // 16 MB
  _Float16* Kh    = (_Float16*)take((size_t)Rows * Ee * 2); // 16 MB
  _Float16* Vt    = (_Float16*)take((size_t)Rows * Ee * 2); // 16 MB
  _Float16* Og    = (_Float16*)take((size_t)Rows * Ee * 2); // 16 MB
  float*    msk   = (float*)take((size_t)Bb * Tt * 4);      // 32 KB

  k_mask<<<(Bb * Tt + 255) / 256, 256, 0, stream>>>(ids, msk, Bb * Tt);
  k_wt<<<dim3(Ee / 256, E3), 256, 0, stream>>>(w_qkv, wqkvT, Ee, E3);
  k_wt<<<dim3(Ee / 256, Ee), 256, 0, stream>>>(w_out, woutT, Ee, Ee);
  k_qkv_gemm<<<dim3(Rows / 128, E3 / 128), 256, 0, stream>>>(x, wqkvT, Qh, Kh, Vt);
  k_attn<<<dim3(Bb * Hh, Tt / 128), 256, 0, stream>>>(Qh, Kh, Vt, msk, Og);
  k_out_gemm<<<dim3(Rows / 128, Ee / 128), 256, 0, stream>>>(Og, woutT, b_out, out);
}
